// HybridTransformerNetwork_16458314678483
// MI455X (gfx1250) — matrix operand order repaired and checked by reading, not compiled, not run on hardware
//
#include <hip/hip_runtime.h>
#include <hip/hip_bf16.h>
#include <math.h>

typedef __attribute__((ext_vector_type(16))) __bf16 v16bf;
typedef __attribute__((ext_vector_type(4)))  __bf16 v4bf;
typedef __attribute__((ext_vector_type(8)))  float  v8f;

union FragU { v16bf v; uint4 q[2]; };

__device__ __forceinline__ int imin(int a, int b) { return a < b ? a : b; }

// float atomic max via signed/unsigned integer ordering trick (valid incl. -inf init)
__device__ __forceinline__ void atomicMaxF(float* addr, float v) {
  if (v >= 0.0f) atomicMax((int*)addr, __float_as_int(v));
  else           atomicMin((unsigned int*)addr, __float_as_uint(v));
}

// ------------------------- encoder: h = x @ enc_W + enc_b -------------------------
__global__ void encoder_kernel(const float* __restrict__ x, const float* __restrict__ W,
                               const float* __restrict__ b, float* __restrict__ h, int N) {
  int n = blockIdx.x * blockDim.x + threadIdx.x;
  if (n >= N) return;
  float xi[5];
#pragma unroll
  for (int k = 0; k < 5; ++k) xi[k] = x[(size_t)n * 5 + k];
  for (int c = 0; c < 64; ++c) {
    float acc = b[c];
#pragma unroll
    for (int k = 0; k < 5; ++k) acc = fmaf(xi[k], W[k * 64 + c], acc);
    h[(size_t)n * 64 + c] = acc;
  }
}

// ------------ per-layer node prologue: z = relu(LN(h)); init m/den/num ------------
__global__ void node_pre_kernel(const float* __restrict__ h, const float* __restrict__ g,
                                const float* __restrict__ bb, float* __restrict__ z,
                                float* __restrict__ m, float* __restrict__ den,
                                float* __restrict__ num, int N) {
  int n = blockIdx.x * blockDim.x + threadIdx.x;
  if (n >= N) return;
  const float* hr = h + (size_t)n * 64;
  float mean = 0.f, var = 0.f;
  for (int c = 0; c < 64; ++c) mean += hr[c];
  mean *= (1.f / 64.f);
  for (int c = 0; c < 64; ++c) { float d = hr[c] - mean; var += d * d; }
  var *= (1.f / 64.f);
  float rs = rsqrtf(var + 1e-5f);
  for (int c = 0; c < 64; ++c) {
    float zz = (hr[c] - mean) * rs * g[c] + bb[c];
    zz = fmaxf(zz, 0.f);
    size_t idx = (size_t)n * 64 + c;
    z[idx]   = zz;
    m[idx]   = -INFINITY;
    den[idx] = 0.f;
    num[idx] = 0.f;
  }
}

// ----------------- edge pass 1: per-channel segment max of s=msg*t -----------------
__global__ void edge_max_kernel(const int* __restrict__ ei, const float* __restrict__ z,
                                const float* __restrict__ tparam, float* __restrict__ m,
                                int E) {
  int idx = blockIdx.x * blockDim.x + threadIdx.x;
  int e = idx >> 2;
  if (e >= E) return;
  int c0 = (idx & 3) * 16;
  int src = ei[e];
  int dst = ei[E + e];
  float t = tparam[0];
  const float* zr = z + (size_t)src * 64 + c0;
  float* mr = m + (size_t)dst * 64 + c0;
#pragma unroll
  for (int c = 0; c < 16; ++c) {
    float msg = fmaxf(zr[c], 0.f) + 1e-7f;
    atomicMaxF(&mr[c], msg * t);
  }
}

// replace -inf (zero in-degree) with 0, matching jnp.where(isfinite(m), m, 0)
__global__ void fixm_kernel(float* __restrict__ m, long total) {
  long i = (long)blockIdx.x * blockDim.x + threadIdx.x;
  if (i >= total) return;
  float v = m[i];
  m[i] = (v > -1e30f) ? v : 0.f;
}

// --------- edge pass 2: den += exp(s-m[dst]); num += msg*exp(s-m[dst]) -------------
__global__ void edge_sum_kernel(const int* __restrict__ ei, const float* __restrict__ z,
                                const float* __restrict__ tparam, const float* __restrict__ m,
                                float* __restrict__ den, float* __restrict__ num, int E) {
  int idx = blockIdx.x * blockDim.x + threadIdx.x;
  int e = idx >> 2;
  if (e >= E) return;
  int c0 = (idx & 3) * 16;
  int src = ei[e];
  int dst = ei[E + e];
  float t = tparam[0];
  const float* zr = z + (size_t)src * 64 + c0;
  const float* mr = m + (size_t)dst * 64 + c0;
  float* dr = den + (size_t)dst * 64 + c0;
  float* nr = num + (size_t)dst * 64 + c0;
#pragma unroll
  for (int c = 0; c < 16; ++c) {
    float msg = fmaxf(zr[c], 0.f) + 1e-7f;
    float ev = __expf(msg * t - mr[c]);
    atomicAdd(&dr[c], ev);
    atomicAdd(&nr[c], msg * ev);
  }
}

// u = z + num / max(den, 1e-16), emitted directly as bf16 (GEMM A-operand precision)
__global__ void node_u_bf16_kernel(const float* __restrict__ z, const float* __restrict__ num,
                                   const float* __restrict__ den, __bf16* __restrict__ ubf,
                                   long total) {
  long i = (long)blockIdx.x * blockDim.x + threadIdx.x;
  if (i >= total) return;
  ubf[i] = (__bf16)(z[i] + num[i] / fmaxf(den[i], 1e-16f));
}

// row-wise LN then ReLU, reading f32, emitting bf16 (GENConv MLP inner norm)
__global__ void ln_relu_bf16_kernel(const float* __restrict__ y, const float* __restrict__ g,
                                    const float* __restrict__ b, __bf16* __restrict__ ybf,
                                    int rows, int dim) {
  int r = blockIdx.x * blockDim.x + threadIdx.x;
  if (r >= rows) return;
  const float* yr = y + (size_t)r * dim;
  __bf16* orow = ybf + (size_t)r * dim;
  float mean = 0.f, var = 0.f;
  for (int c = 0; c < dim; ++c) mean += yr[c];
  mean /= (float)dim;
  for (int c = 0; c < dim; ++c) { float d = yr[c] - mean; var += d * d; }
  var /= (float)dim;
  float rs = rsqrtf(var + 1e-5f);
  for (int c = 0; c < dim; ++c)
    orow[c] = (__bf16)fmaxf((yr[c] - mean) * rs * g[c] + b[c], 0.f);
}

// t = LN(t + delta) (post-norm residual)
__global__ void add_ln_kernel(float* __restrict__ t, const float* __restrict__ delta,
                              const float* __restrict__ g, const float* __restrict__ b,
                              int rows, int dim) {
  int r = blockIdx.x * blockDim.x + threadIdx.x;
  if (r >= rows) return;
  float* tr = t + (size_t)r * dim;
  const float* dr = delta + (size_t)r * dim;
  float mean = 0.f, var = 0.f;
  for (int c = 0; c < dim; ++c) mean += tr[c] + dr[c];
  mean /= (float)dim;
  for (int c = 0; c < dim; ++c) { float d = tr[c] + dr[c] - mean; var += d * d; }
  var /= (float)dim;
  float rs = rsqrtf(var + 1e-5f);
  for (int c = 0; c < dim; ++c)
    tr[c] = ((tr[c] + dr[c]) - mean) * rs * g[c] + b[c];
}

__global__ void softmax_rows_kernel(float* __restrict__ s, int rows, int dim) {
  int r = blockIdx.x * blockDim.x + threadIdx.x;
  if (r >= rows) return;
  float* p = s + (size_t)r * dim;
  float mx = -INFINITY;
  for (int i = 0; i < dim; ++i) mx = fmaxf(mx, p[i]);
  float sum = 0.f;
  for (int i = 0; i < dim; ++i) { float e = __expf(p[i] - mx); p[i] = e; sum += e; }
  float inv = 1.f / sum;
  for (int i = 0; i < dim; ++i) p[i] *= inv;
}

__global__ void zero_kernel(float* __restrict__ p, long total) {
  long i = (long)blockIdx.x * blockDim.x + threadIdx.x;
  if (i < total) p[i] = 0.f;
}

__global__ void pool_kernel(const float* __restrict__ h, const int* __restrict__ batch,
                            float* __restrict__ pooled, long total) {
  long idx = (long)blockIdx.x * blockDim.x + threadIdx.x;
  if (idx >= total) return;
  long n = idx >> 6;
  int c = (int)(idx & 63);
  atomicAdd(&pooled[(size_t)batch[n] * 64 + c], h[(size_t)n * 64 + c]);
}

// variable-embed MLP 19->16->32->64 and concat with pooled into t[B,128]
__global__ void var_concat_kernel(const float* __restrict__ var, const float* __restrict__ pooled,
                                  const float* __restrict__ w1, const float* __restrict__ b1,
                                  const float* __restrict__ w2, const float* __restrict__ b2,
                                  const float* __restrict__ w3, const float* __restrict__ b3,
                                  float* __restrict__ t, int B) {
  int r = blockIdx.x * blockDim.x + threadIdx.x;
  if (r >= B) return;
  float a1[16], a2[32];
  for (int j = 0; j < 16; ++j) {
    float acc = b1[j];
    for (int k = 0; k < 19; ++k) acc = fmaf(var[(size_t)r * 19 + k], w1[k * 16 + j], acc);
    a1[j] = fmaxf(acc, 0.f);
  }
  for (int j = 0; j < 32; ++j) {
    float acc = b2[j];
    for (int k = 0; k < 16; ++k) acc = fmaf(a1[k], w2[k * 32 + j], acc);
    a2[j] = fmaxf(acc, 0.f);
  }
  for (int j = 0; j < 64; ++j) {
    float acc = b3[j];
    for (int k = 0; k < 32; ++k) acc = fmaf(a2[k], w3[k * 64 + j], acc);
    t[(size_t)r * 128 + 64 + j] = acc;
  }
  for (int c = 0; c < 64; ++c) t[(size_t)r * 128 + c] = pooled[(size_t)r * 64 + c];
}

// ------------------------------------------------------------------
// LDS-staged bf16 WMMA GEMM.
//   C[m,n] (= or +=) act(alpha * sum_k A(m,k)*B(k,n) + bias[n])
//   A(m,k): abf=0 -> f32 at A[aoff + y*astride + m*lda + k]
//           abf=1 -> bf16 at ((bf16*)A)[...]; staged via global_load_async_to_lds_b128
//                    (requires K%32==0; tail M rows clamped, masked at epilogue)
//   B(k,n) = f32 at B[boff + y*bstride + k*ldbk + n*ldbn]   (y = blockIdx.y, heads)
// Block: 128 threads / 4 waves, 64x64 macro-tile, K stepped by 32.
// Fragments: unconditional 128-bit LDS loads, contiguous k = 16*half + s for A and B; all four
// B fragments are hoisted before the WMMA chain so ds_loads overlap WMMA issue.
// LDS row stride = 40 bf16 (80B) -> 16 lane rows land on distinct bank groups.
// ------------------------------------------------------------------
#define LSTR 40
__global__ __launch_bounds__(128)
void wmma_gemm_tiled(const float* __restrict__ A, long aoff, long astride, int lda, int abf,
                     const float* __restrict__ B, long boff, long bstride, int ldbk, int ldbn,
                     const float* __restrict__ bias,
                     float* __restrict__ C, long coff, long cstride, int ldc,
                     int M, int Nmat, int K, float alpha, int act, int accum) {
  __shared__ __bf16 Alds[64 * LSTR];
  __shared__ __bf16 Blds[64 * LSTR];   // transposed: [n][k]
  aoff += (long)blockIdx.y * astride;
  boff += (long)blockIdx.y * bstride;
  coff += (long)blockIdx.y * cstride;
  int tilesN = (Nmat + 63) >> 6;
  int bm = (int)(blockIdx.x / tilesN) * 64;
  int bn = (int)(blockIdx.x % tilesN) * 64;
  int tid  = threadIdx.x;
  int wave = tid >> 5;
  int lane = tid & 31;
  int half = lane >> 4;
  int lrow = lane & 15;
  bool alA = (((unsigned)lda | (unsigned)(aoff & 3)) & 3u) == 0u;
  bool alB = (ldbn == 1) && ((((unsigned)ldbk | (unsigned)(boff & 3)) & 3u) == 0u);
  v8f acc0 = {}, acc1 = {}, acc2 = {}, acc3 = {};
  for (int k0 = 0; k0 < K; k0 += 32) {
    __syncthreads();
    if (abf) {
      // ---- stage A via async DMA: 64 rows x 32 bf16 = 256 x 16B chunks ----
      const __bf16* Ab = (const __bf16*)A;
#pragma unroll
      for (int it = 0; it < 2; ++it) {
        int chunk = it * 128 + tid;          // 0..255
        int r  = chunk >> 2;                 // LDS row 0..63
        int cc = chunk & 3;                  // 16B chunk within row
        int mg = imin(bm + r, M - 1);        // clamp: garbage rows masked at store
        const __bf16* gp = Ab + aoff + (long)mg * lda + k0 + cc * 8;
        unsigned lo = (unsigned)(unsigned long long)&Alds[r * LSTR + cc * 8];
        asm volatile("global_load_async_to_lds_b128 %0, %1, off"
                     :: "v"(lo), "v"(gp) : "memory");
      }
    } else {
      // ---- stage A (f32 -> bf16): 64 rows x 32 k ----
#pragma unroll
      for (int it = 0; it < 4; ++it) {
        int i = it * 512 + tid * 4;
        int r = i >> 5, kk = i & 31;
        int mg = bm + r;
        int ka = k0 + kk;
        v4bf p;
        if (alA && mg < M && ka + 3 < K) {
          float4 f = *(const float4*)(A + aoff + (long)mg * lda + ka);
          p[0] = (__bf16)f.x; p[1] = (__bf16)f.y; p[2] = (__bf16)f.z; p[3] = (__bf16)f.w;
        } else {
#pragma unroll
          for (int j = 0; j < 4; ++j) {
            float v = 0.f;
            if (mg < M && ka + j < K) v = A[aoff + (long)mg * lda + ka + j];
            p[j] = (__bf16)v;
          }
        }
        *(v4bf*)&Alds[r * LSTR + kk] = p;
      }
    }
    // ---- stage B (transposed into LDS): read k-major, write [n][k] ----
#pragma unroll
    for (int it = 0; it < 4; ++it) {
      int i = it * 512 + tid * 4;
      int kk = i >> 6, nn = i & 63;
      int kb = k0 + kk;
      int ng = bn + nn;
      float4 f;
      if (alB && kb < K && ng + 3 < Nmat) {
        f = *(const float4*)(B + boff + (long)kb * ldbk + ng);
      } else {
        float* fp = (float*)&f;
#pragma unroll
        for (int j = 0; j < 4; ++j) {
          float v = 0.f;
          if (kb < K && ng + j < Nmat) v = B[boff + (long)kb * ldbk + (long)(ng + j) * ldbn];
          fp[j] = v;
        }
      }
      Blds[(nn + 0) * LSTR + kk] = (__bf16)f.x;
      Blds[(nn + 1) * LSTR + kk] = (__bf16)f.y;
      Blds[(nn + 2) * LSTR + kk] = (__bf16)f.z;
      Blds[(nn + 3) * LSTR + kk] = (__bf16)f.w;
    }
    if (abf) asm volatile("s_wait_asynccnt 0x0" ::: "memory");
    __syncthreads();
    // ---- fragments: issue ALL ds_loads first so waits stagger across the WMMA chain
    FragU fa, fb0, fb1, fb2, fb3;
    {
      const uint4* pa = (const uint4*)&Alds[(wave * 16 + lrow) * LSTR];
      fa.q[0] = pa[half * 2];      // k bytes [half*32, +16)
      fa.q[1] = pa[half * 2 + 1];  // k bytes [half*32 + 16, +16)
      const uint4* pb0 = (const uint4*)&Blds[(0 * 16 + lrow) * LSTR];
      const uint4* pb1 = (const uint4*)&Blds[(1 * 16 + lrow) * LSTR];
      const uint4* pb2 = (const uint4*)&Blds[(2 * 16 + lrow) * LSTR];
      const uint4* pb3 = (const uint4*)&Blds[(3 * 16 + lrow) * LSTR];
      fb0.q[0] = pb0[half * 2]; fb0.q[1] = pb0[half * 2 + 1];
      fb1.q[0] = pb1[half * 2]; fb1.q[1] = pb1[half * 2 + 1];
      fb2.q[0] = pb2[half * 2]; fb2.q[1] = pb2[half * 2 + 1];
      fb3.q[0] = pb3[half * 2]; fb3.q[1] = pb3[half * 2 + 1];
    }
    acc0 = __builtin_amdgcn_wmma_f32_16x16x32_bf16(false, fa.v, false, fb0.v, (short)0, acc0, false, false);
    acc1 = __builtin_amdgcn_wmma_f32_16x16x32_bf16(false, fa.v, false, fb1.v, (short)0, acc1, false, false);
    acc2 = __builtin_amdgcn_wmma_f32_16x16x32_bf16(false, fa.v, false, fb2.v, (short)0, acc2, false, false);
    acc3 = __builtin_amdgcn_wmma_f32_16x16x32_bf16(false, fa.v, false, fb3.v, (short)0, acc3, false, false);
  }
  // ---- epilogue: C/D layout lane(n=lrow), VGPR r -> m = r + 8*half ----
  int mBase = bm + wave * 16 + half * 8;
#pragma unroll
  for (int j = 0; j < 4; ++j) {
    v8f* ac = (j == 0) ? &acc0 : (j == 1) ? &acc1 : (j == 2) ? &acc2 : &acc3;
    int nC = bn + j * 16 + lrow;
#pragma unroll
    for (int r = 0; r < 8; ++r) {
      int mC = mBase + r;
      if (mC < M && nC < Nmat) {
        float v = (*ac)[r] * alpha;
        if (bias) v += bias[nC];
        if (act == 1) v = 0.5f * v * (1.0f + erff(v * 0.70710678118f));  // exact GELU
        else if (act == 2) v = fmaxf(v, 0.f);
        long idx = coff + (long)mC * ldc + nC;
        if (accum) v += C[idx];
        C[idx] = v;
      }
    }
  }
}

extern "C" void kernel_launch(void* const* d_in, const int* in_sizes, int n_in,
                              void* d_out, int out_size, void* d_ws, size_t ws_size,
                              hipStream_t stream) {
  (void)n_in; (void)out_size; (void)ws_size;
  const float* x       = (const float*)d_in[0];
  const int*   ei      = (const int*)d_in[1];
  const int*   batch   = (const int*)d_in[2];
  const float* var_d   = (const float*)d_in[3];
  const float* enc_W   = (const float*)d_in[4];
  const float* enc_b   = (const float*)d_in[5];
  const float* ln_g    = (const float*)d_in[6];
  const float* ln_b    = (const float*)d_in[7];
  const float* t_param = (const float*)d_in[8];
  const float* mlp1_W  = (const float*)d_in[9];
  const float* mlp1_b  = (const float*)d_in[10];
  const float* mlpn_g  = (const float*)d_in[11];
  const float* mlpn_b  = (const float*)d_in[12];
  const float* mlp2_W  = (const float*)d_in[13];
  const float* mlp2_b  = (const float*)d_in[14];
  const float* v1_W = (const float*)d_in[15]; const float* v1_b = (const float*)d_in[16];
  const float* v2_W = (const float*)d_in[17]; const float* v2_b = (const float*)d_in[18];
  const float* v3_W = (const float*)d_in[19]; const float* v3_b = (const float*)d_in[20];
  const float* qkv_W = (const float*)d_in[21]; const float* qkv_b = (const float*)d_in[22];
  const float* out_W = (const float*)d_in[23]; const float* out_b = (const float*)d_in[24];
  const float* ln1_g = (const float*)d_in[25]; const float* ln1_b = (const float*)d_in[26];
  const float* ff1_W = (const float*)d_in[27]; const float* ff1_b = (const float*)d_in[28];
  const float* ff2_W = (const float*)d_in[29]; const float* ff2_b = (const float*)d_in[30];
  const float* ln2_g = (const float*)d_in[31]; const float* ln2_b = (const float*)d_in[32];
  const float* head_W = (const float*)d_in[33]; const float* head_b = (const float*)d_in[34];

  const int N = in_sizes[0] / 5;
  const int E = in_sizes[1] / 2;
  const int B = in_sizes[3] / 19;

  float* ws = (float*)d_ws;
  size_t o = 0;
  float* h   = ws + o; o += (size_t)N * 64;
  float* z   = ws + o; o += (size_t)N * 64;
  float* num = ws + o; o += (size_t)N * 64;   // reused as bf16 y after aggregation
  float* y   = ws + o; o += (size_t)N * 128;  // MLP hidden (f32); aliases m/den below
  float* ubf_f = ws + o; o += (size_t)N * 32; // bf16 u = z + agg  (N*64 bf16)
  float* m   = y;                             // segment max (dead before y written)
  float* den = y + (size_t)N * 64;            // segment denom (dead before y written)
  __bf16* ubf = (__bf16*)ubf_f;
  __bf16* ybf = (__bf16*)num;                 // N*128 bf16 == N*64 f32 region
  float* pooled = ws + o; o += (size_t)B * 64;
  float* t      = ws + o; o += (size_t)B * 128;
  float* qkvb   = ws + o; o += (size_t)B * 384;
  float* scores = ws + o; o += (size_t)8 * B * B;
  float* obuf   = ws + o; o += (size_t)B * 128;
  float* tmp    = ws + o; o += (size_t)B * 128;

  auto gemm = [&](const float* A, long aoff, long astride, int lda, int abf,
                  const float* Bm, long boff, long bstride, int ldbk, int ldbn,
                  const float* bias, float* C, long coff, long cstride, int ldc,
                  int M_, int N_, int K_, int nbatch, float alpha, int act, int accum) {
    int tiles = ((M_ + 63) / 64) * ((N_ + 63) / 64);
    hipLaunchKernelGGL(wmma_gemm_tiled, dim3(tiles, nbatch), dim3(128), 0, stream,
                       A, aoff, astride, lda, abf, Bm, boff, bstride, ldbk, ldbn, bias,
                       C, coff, cstride, ldc, M_, N_, K_, alpha, act, accum);
  };

  const long NE64 = (long)N * 64;

  encoder_kernel<<<(N + 255) / 256, 256, 0, stream>>>(x, enc_W, enc_b, h, N);

  for (int i = 0; i < 6; ++i) {
    node_pre_kernel<<<(N + 255) / 256, 256, 0, stream>>>(
        h, ln_g + i * 64, ln_b + i * 64, z, m, den, num, N);
    long ethreads = (long)E * 4;
    edge_max_kernel<<<(ethreads + 255) / 256, 256, 0, stream>>>(ei, z, t_param + i, m, E);
    fixm_kernel<<<(NE64 + 255) / 256, 256, 0, stream>>>(m, NE64);
    edge_sum_kernel<<<(ethreads + 255) / 256, 256, 0, stream>>>(
        ei, z, t_param + i, m, den, num, E);
    node_u_bf16_kernel<<<(NE64 + 255) / 256, 256, 0, stream>>>(z, num, den, ubf, NE64);
    // y = u @ mlp1_W [64,128] + b1   (bf16 A via async-to-LDS)
    gemm((const float*)ubf, 0, 0, 64, 1, mlp1_W + (long)i * 64 * 128, 0, 0, 128, 1,
         mlp1_b + i * 128, y, 0, 0, 128, N, 128, 64, 1, 1.f, 0, 0);
    ln_relu_bf16_kernel<<<(N + 255) / 256, 256, 0, stream>>>(
        y, mlpn_g + i * 128, mlpn_b + i * 128, ybf, N, 128);
    // h += ybf @ mlp2_W [128,64] + b2   (bf16 A via async-to-LDS, accumulate)
    gemm((const float*)ybf, 0, 0, 128, 1, mlp2_W + (long)i * 128 * 64, 0, 0, 64, 1,
         mlp2_b + i * 64, h, 0, 0, 64, N, 64, 128, 1, 1.f, 0, 1);
  }

  zero_kernel<<<((long)B * 64 + 255) / 256, 256, 0, stream>>>(pooled, (long)B * 64);
  pool_kernel<<<(NE64 + 255) / 256, 256, 0, stream>>>(h, batch, pooled, NE64);
  var_concat_kernel<<<(B + 255) / 256, 256, 0, stream>>>(
      var_d, pooled, v1_W, v1_b, v2_W, v2_b, v3_W, v3_b, t, B);

  for (int i = 0; i < 6; ++i) {
    // qkv = t @ qkv_W + b  -> [B, 384]
    gemm(t, 0, 0, 128, 0, qkv_W + (long)i * 128 * 384, 0, 0, 384, 1, qkv_b + i * 384,
         qkvb, 0, 0, 384, B, 384, 128, 1, 1.f, 0, 0);
    // scores[hd] = (q_hd @ k_hd^T) / sqrt(16)   (batched over 8 heads via grid.y)
    gemm(qkvb, 0, 16, 384, 0, qkvb, 128, 16, 1, 384, nullptr,
         scores, 0, (long)B * B, B, B, B, 16, 8, 0.25f, 0, 0);
    softmax_rows_kernel<<<(8 * B + 255) / 256, 256, 0, stream>>>(scores, 8 * B, B);
    // o[:, hd*16:+16] = att_hd @ v_hd   (batched over heads)
    gemm(scores, 0, (long)B * B, B, 0, qkvb, 256, 16, 384, 1, nullptr,
         obuf, 0, 16, 128, B, 16, B, 8, 1.f, 0, 0);
    gemm(obuf, 0, 0, 128, 0, out_W + (long)i * 128 * 128, 0, 0, 128, 1, out_b + i * 128,
         tmp, 0, 0, 128, B, 128, 128, 1, 1.f, 0, 0);
    add_ln_kernel<<<(B + 255) / 256, 256, 0, stream>>>(
        t, tmp, ln1_g + i * 128, ln1_b + i * 128, B, 128);
    gemm(t, 0, 0, 128, 0, ff1_W + (long)i * 128 * 128, 0, 0, 128, 1, ff1_b + i * 128,
         obuf, 0, 0, 128, B, 128, 128, 1, 1.f, 1 /*gelu*/, 0);
    gemm(obuf, 0, 0, 128, 0, ff2_W + (long)i * 128 * 128, 0, 0, 128, 1, ff2_b + i * 128,
         tmp, 0, 0, 128, B, 128, 128, 1, 1.f, 0, 0);
    add_ln_kernel<<<(B + 255) / 256, 256, 0, stream>>>(
        t, tmp, ln2_g + i * 128, ln2_b + i * 128, B, 128);
  }

  // head: out = t @ head_W [128,2] + head_b
  gemm(t, 0, 0, 128, 0, head_W, 0, 0, 2, 1, head_b, (float*)d_out, 0, 0, 2, B, 2, 128,
       1, 1.f, 0, 0);
}
